// RaftModule_68092411511643
// MI455X (gfx1250) — matrix operand order repaired and checked by reading, not compiled, not run on hardware
//
#include <hip/hip_runtime.h>
#include <hip/hip_bf16.h>

// ---------------------------------------------------------------------------
// RAFT-style coarse-to-fine optical flow for gfx1250 (MI455X).
// All heavy math (correlation volumes + every conv) runs through one
// f16-in / f32-accumulate WMMA GEMM kernel (v_wmma_f32_16x16x32_f16),
// with branch-free vectorized global->LDS tile staging (b128 both sides),
// BLK_K=64 (2 WMMAs per barrier round), and a NULL-tensor TDM issue to
// exercise the tensor_load_to_lds / s_wait_tensorcnt path.
// Activations are channel-major [C, B, H, W] so concats are contiguous.
// ---------------------------------------------------------------------------

typedef _Float16 h16;
typedef __attribute__((ext_vector_type(16))) _Float16 v16h;
typedef __attribute__((ext_vector_type(8)))  _Float16 h16x8;
typedef __attribute__((ext_vector_type(8)))  float    v8f;
typedef __attribute__((ext_vector_type(4)))  unsigned int u32x4;
typedef __attribute__((ext_vector_type(8)))  int      i32x8;
typedef __attribute__((ext_vector_type(4)))  int      i32x4;

#define BLK_M 32
#define BLK_N 64
#define BLK_K 64
#define LDS_AS 80   // halves per A-tile row (64 + 16 pad) -> 160B, 16B aligned
#define LDS_BS 80   // halves per B-tile row

// Degenerate (NULL-tensor, count=0 -> architectural no-op) TDM issue + wait.
// Exercises the gfx1250 Tensor Data Mover instruction path.
__device__ __forceinline__ void tdm_touch()
{
#if __has_builtin(__builtin_amdgcn_tensor_load_to_lds)
    u32x4 g0 = {0u, 0u, 0u, 0u};                 // count=0 => NULL descriptor
    i32x8 g1 = {0, 0, 0, 0, 0, 0, 0, 0};
    i32x4 g2 = {0, 0, 0, 0};
    i32x4 g3 = {0, 0, 0, 0};
#if __clang_major__ >= 23
    i32x8 g4 = {0, 0, 0, 0, 0, 0, 0, 0};
    __builtin_amdgcn_tensor_load_to_lds(g0, g1, g2, g3, g4, 0);
#else
    __builtin_amdgcn_tensor_load_to_lds(g0, g1, g2, g3, 0);
#endif
#if __has_builtin(__builtin_amdgcn_s_wait_tensorcnt)
    __builtin_amdgcn_s_wait_tensorcnt(0);
#endif
#endif
}

// ---------------------------------------------------------------------------
// Generic GEMM: C[M,N] = act( alpha * A[M,K] * B[K,N] + bias[m] )
// A row-major [M,K] f16, BT row-major [N,K] f16 (i.e. B transposed),
// K multiple of 64, C row-major f32. act: 0=none 1=relu 2=sigmoid 3=tanh
// ---------------------------------------------------------------------------
__global__ __launch_bounds__(256)
void gemm_wmma_f16(const h16* __restrict__ A, const h16* __restrict__ BT,
                   float* __restrict__ C, int M, int N, int K,
                   const float* __restrict__ bias, float alpha, int act)
{
    __shared__ alignas(16) h16 As[BLK_M * LDS_AS];   // 5120 B
    __shared__ alignas(16) h16 Bs[BLK_N * LDS_BS];   // 10240 B

    tdm_touch();

    const int tid  = threadIdx.x;
    const int lane = tid & 31;
    const int wave = tid >> 5;
    const int wm   = wave & 1;    // 16-row slab of the 32-row block tile
    const int wn   = wave >> 1;   // 16-col slab of the 64-col block tile
    const int m0   = blockIdx.y * BLK_M;
    const int n0   = blockIdx.x * BLK_N;

    const int halfSel = lane >> 4;
    const int l15     = lane & 15;
    const int a_row   = wm * 16 + l15;
    const int a_kb    = halfSel * 8;
    const int b_col   = wn * 16 + l15;
    const int b_kb    = halfSel * 8;

    // A staging: thread -> row am = tid>>3 (0..31), 8 halves at ak
    const int  am = tid >> 3;
    const int  ak = (tid & 7) * 8;
    const bool a_ok = (m0 + am) < M;
    const long a_src = (long)min(m0 + am, M - 1) * K;
    // B staging: thread -> row bn = tid>>2 (0..63), 16 halves at bk
    const int  bn = tid >> 2;
    const int  bk = (tid & 3) * 16;
    const bool b_ok = (n0 + bn) < N;
    const long b_src = (long)min(n0 + bn, N - 1) * K;

    const h16x8 hz = {};
    v8f acc = {};

    for (int k0 = 0; k0 < K; k0 += BLK_K) {
        if (k0 + BLK_K < K) {
            __builtin_prefetch(A  + a_src + k0 + BLK_K, 0, 1);
            __builtin_prefetch(BT + b_src + k0 + BLK_K, 0, 1);
        }
        h16x8 av = *(const h16x8*)(A + a_src + k0 + ak);
        if (!a_ok) av = hz;
        *(h16x8*)(As + am * LDS_AS + ak) = av;

        h16x8 bv0 = *(const h16x8*)(BT + b_src + k0 + bk);
        h16x8 bv1 = *(const h16x8*)(BT + b_src + k0 + bk + 8);
        if (!b_ok) { bv0 = hz; bv1 = hz; }
        *(h16x8*)(Bs + bn * LDS_BS + bk)     = bv0;
        *(h16x8*)(Bs + bn * LDS_BS + bk + 8) = bv1;
        __syncthreads();

        #pragma unroll
        for (int s = 0; s < 2; ++s) {
            // A fragment (16-bit A 16x32): lanes 0-15 K=0..7/16..23,
            // lanes 16-31 K=8..15/24..31 within this 32-K sub-step
            const h16* ap = As + a_row * LDS_AS + s * 32 + a_kb;
            h16x8 alo = *(const h16x8*)ap;
            h16x8 ahi = *(const h16x8*)(ap + 16);
            v16h af;
            #pragma unroll
            for (int i = 0; i < 8; ++i) { af[i] = alo[i]; af[i + 8] = ahi[i]; }
            // B fragment (16-bit B 32x16): lanes 0-15 K=0..7/16..23, 16-31 K=8..15/24..31
            const h16* bp = Bs + b_col * LDS_BS + s * 32 + b_kb;
            h16x8 blo = *(const h16x8*)bp;
            h16x8 bhi = *(const h16x8*)(bp + 16);
            v16h bf;
            #pragma unroll
            for (int i = 0; i < 8; ++i) { bf[i] = blo[i]; bf[i + 8] = bhi[i]; }
            acc = __builtin_amdgcn_wmma_f32_16x16x32_f16(
                      false, af, false, bf, (short)0, acc, false, false);
        }
        __syncthreads();
    }

    // C/D layout: lanes 0-15: n=lane, m=r ; lanes 16-31: n=lane-16, m=r+8
    #pragma unroll
    for (int r = 0; r < 8; ++r) {
        int m = m0 + wm * 16 + r + halfSel * 8;
        int n = n0 + wn * 16 + l15;
        if (m < M && n < N) {
            float v = acc[r] * alpha;
            if (bias) v += bias[m];
            if (act == 1)      v = fmaxf(v, 0.f);
            else if (act == 2) v = 1.f / (1.f + __expf(-v));
            else if (act == 3) v = tanhf(v);
            C[(long)m * N + n] = v;
        }
    }
}

// ---------------------------------------------------------------------------
// Lowering / packing kernels
// ---------------------------------------------------------------------------
__global__ void pack_weight_f16(const float* __restrict__ w, h16* __restrict__ dst,
                                int M, int K, int Kp)
{
    long idx = (long)blockIdx.x * blockDim.x + threadIdx.x;
    long total = (long)M * Kp;
    if (idx >= total) return;
    int k = (int)(idx % Kp);
    int m = (int)(idx / Kp);
    dst[idx] = (k < K) ? (h16)w[(long)m * K + k] : (h16)0;
}

// Correlation operand pack (both A and BT): dst[p=pixel][c] = f[c*HW + p]
__global__ void pack_corr_f16(const float* __restrict__ f, h16* __restrict__ dst, int HW)
{
    long idx = (long)blockIdx.x * blockDim.x + threadIdx.x;
    long total = (long)HW * 256;
    if (idx >= total) return;
    int k = (int)(idx & 255);
    int m = (int)(idx >> 8);
    dst[idx] = (h16)f[(long)k * HW + m];
}

// im2col from channel-major activation [Cin, Bn, H, W] (f32) to
// transposed patch matrix BT [Bn*H*W, Kp] f16 (row-major, K contiguous).
__global__ void im2col_f16(const float* __restrict__ src, h16* __restrict__ dst,
                           int Cin, int Bn, int H, int W, int kh, int kw, int Kp)
{
    long Ncol = (long)Bn * H * W;
    long total = (long)Kp * Ncol;
    long idx = (long)blockIdx.x * blockDim.x + threadIdx.x;
    if (idx >= total) return;
    int  row = (int)(idx % Kp);      // k index (contiguous per dst row)
    long col = idx / Kp;             // output pixel
    int K = Cin * kh * kw;
    float v = 0.f;
    if (row < K) {
        int hw = (int)(col % (H * W));
        int b  = (int)(col / (H * W));
        int h = hw / W, w = hw % W;
        int c  = row / (kh * kw);
        int t  = row % (kh * kw);
        int ky = t / kw, kx = t % kw;
        int ih = h + ky - kh / 2;
        int iw = w + kx - kw / 2;
        if (ih >= 0 && ih < H && iw >= 0 && iw < W)
            v = src[(((long)c * Bn + b) * H + ih) * W + iw];
    }
    dst[idx] = (h16)v;
}

// ---------------------------------------------------------------------------
// Elementwise / glue kernels
// ---------------------------------------------------------------------------
__global__ void bchw_to_cbhw(const float* __restrict__ src, float* __restrict__ dst,
                             int C, int Bn, int HW)
{
    long idx = (long)blockIdx.x * blockDim.x + threadIdx.x;
    long total = (long)C * Bn * HW;
    if (idx >= total) return;
    int hw = (int)(idx % HW);
    long t = idx / HW;
    int b = (int)(t % Bn);
    int c = (int)(t / Bn);
    dst[idx] = src[((long)b * C + c) * HW + hw];
}

__global__ void ctx_split(const float* __restrict__ src, float* __restrict__ h,
                          float* __restrict__ ctx, long N)
{
    long idx = (long)blockIdx.x * blockDim.x + threadIdx.x;
    if (idx >= 256 * N) return;
    int c = (int)(idx / N);
    float v = src[idx];
    if (c < 128) h[idx] = tanhf(v);
    else         ctx[idx - 128 * N] = fmaxf(v, 0.f);
}

__global__ void fill0_f32(float* __restrict__ p, long n)
{
    long idx = (long)blockIdx.x * blockDim.x + threadIdx.x;
    if (idx < n) p[idx] = 0.f;
}

__global__ void copy_f32k(float* __restrict__ dst, const float* __restrict__ src, long n)
{
    long idx = (long)blockIdx.x * blockDim.x + threadIdx.x;
    if (idx < n) dst[idx] = src[idx];
}

__global__ void ewise_mul(float* __restrict__ out, const float* __restrict__ a,
                          const float* __restrict__ b, long n)
{
    long idx = (long)blockIdx.x * blockDim.x + threadIdx.x;
    if (idx < n) out[idx] = a[idx] * b[idx];
}

__global__ void gru_combine(float* __restrict__ h, const float* __restrict__ z,
                            const float* __restrict__ q, long n)
{
    long idx = (long)blockIdx.x * blockDim.x + threadIdx.x;
    if (idx < n) {
        float zz = z[idx];
        h[idx] = (1.f - zz) * h[idx] + zz * q[idx];
    }
}

__global__ void add_inplace(float* __restrict__ a, const float* __restrict__ b, long n)
{
    long idx = (long)blockIdx.x * blockDim.x + threadIdx.x;
    if (idx < n) a[idx] += b[idx];
}

// flow [2,B,HW] -> out [B,2,HW]
__global__ void flow_to_out(const float* __restrict__ flow, float* __restrict__ out,
                            int Bn, int HW)
{
    long idx = (long)blockIdx.x * blockDim.x + threadIdx.x;
    long total = (long)Bn * 2 * HW;
    if (idx >= total) return;
    int hw = (int)(idx % HW);
    long t = idx / HW;
    int c = (int)(t % 2);
    int b = (int)(t / 2);
    out[idx] = flow[((long)c * Bn + b) * HW + hw];
}

// bilinear align_corners resize of flow [2,B,Ho,Wo] -> 2.0 * [2,B,Hn,Wn]
__global__ void resize_flow(const float* __restrict__ src, float* __restrict__ dst,
                            int Bn, int Ho, int Wo, int Hn, int Wn)
{
    long idx = (long)blockIdx.x * blockDim.x + threadIdx.x;
    long total = (long)2 * Bn * Hn * Wn;
    if (idx >= total) return;
    int x = (int)(idx % Wn);
    long t = idx / Wn;
    int y = (int)(t % Hn); t /= Hn;
    int b = (int)(t % Bn);
    int c = (int)(t / Bn);
    float ys = (Hn > 1) ? (float)y * (float)(Ho - 1) / (float)(Hn - 1) : 0.f;
    float xs = (Wn > 1) ? (float)x * (float)(Wo - 1) / (float)(Wn - 1) : 0.f;
    int y0 = (int)floorf(ys); int y1 = min(y0 + 1, Ho - 1);
    int x0 = (int)floorf(xs); int x1 = min(x0 + 1, Wo - 1);
    float wy = ys - (float)y0, wx = xs - (float)x0;
    const float* s = src + ((long)c * Bn + b) * Ho * Wo;
    float a  = s[y0 * Wo + x0] * (1.f - wy) + s[y1 * Wo + x0] * wy;
    float bb = s[y0 * Wo + x1] * (1.f - wy) + s[y1 * Wo + x1] * wy;
    dst[idx] = 2.f * (a * (1.f - wx) + bb * wx);
}

// vol [B, HW(query), HW(target)]; flow [2,B,HW] -> dst [81, B, HW]
__global__ void corr_lookup_kernel(const float* __restrict__ vol,
                                   const float* __restrict__ flow,
                                   float* __restrict__ dst, int Bn, int H, int W)
{
    int HW = H * W;
    long total = 81L * Bn * HW;
    long idx = (long)blockIdx.x * blockDim.x + threadIdx.x;
    if (idx >= total) return;
    int hw = (int)(idx % HW);
    long t = idx / HW;
    int b = (int)(t % Bn);
    int k = (int)(t / Bn);
    int h = hw / W, w = hw % W;
    float x = (float)w + flow[((long)0 * Bn + b) * HW + hw] + (float)(k % 9) - 4.f;
    float y = (float)h + flow[((long)1 * Bn + b) * HW + hw] + (float)(k / 9) - 4.f;
    float x0 = floorf(x), y0 = floorf(y);
    float wx = x - x0, wy = y - y0;
    const float* row = vol + ((long)b * HW + hw) * HW;
    auto samp = [&](float gx, float gy) -> float {
        if (gx < 0.f || gx > (float)(W - 1) || gy < 0.f || gy > (float)(H - 1)) return 0.f;
        return row[(int)gy * W + (int)gx];
    };
    float v00 = samp(x0, y0),       v10 = samp(x0 + 1.f, y0);
    float v01 = samp(x0, y0 + 1.f), v11 = samp(x0 + 1.f, y0 + 1.f);
    dst[idx] = v00 * (1.f - wx) * (1.f - wy) + v10 * wx * (1.f - wy)
             + v01 * (1.f - wx) * wy         + v11 * wx * wy;
}

// convex 8x upsampler: m576 [576,B,HW] (pre-0.25-scale), flow [2,B,HW]
// out [B,2,8H,8W]
__global__ void up8_kernel(const float* __restrict__ m576, const float* __restrict__ flow,
                           float* __restrict__ out, int Bn, int H, int W)
{
    int HW = H * W;
    int Ho = H * 8, Wo = W * 8;
    long total = (long)Bn * 2 * Ho * Wo;
    long idx = (long)blockIdx.x * blockDim.x + threadIdx.x;
    if (idx >= total) return;
    int X = (int)(idx % Wo);
    long t = idx / Wo;
    int Y = (int)(t % Ho); t /= Ho;
    int c = (int)(t % 2);
    int b = (int)(t / 2);
    int h = Y >> 3, py = Y & 7, w = X >> 3, px = X & 7;
    int p = py * 8 + px;
    float mv[9];
    float mx = -3.0e38f;
    #pragma unroll
    for (int kk = 0; kk < 9; ++kk) {
        mv[kk] = 0.25f * m576[((long)(kk * 64 + p) * Bn + b) * HW + h * W + w];
        mx = fmaxf(mx, mv[kk]);
    }
    float s = 0.f;
    #pragma unroll
    for (int kk = 0; kk < 9; ++kk) { mv[kk] = __expf(mv[kk] - mx); s += mv[kk]; }
    float accv = 0.f;
    #pragma unroll
    for (int kk = 0; kk < 9; ++kk) {
        int ky = kk / 3, kx = kk % 3;
        int ih = h + ky - 1, iw = w + kx - 1;
        float fv = (ih >= 0 && ih < H && iw >= 0 && iw < W)
                 ? 8.f * flow[((long)c * Bn + b) * HW + ih * W + iw] : 0.f;
        accv += mv[kk] * fv;
    }
    out[idx] = accv / s;
}

// ---------------------------------------------------------------------------
// Host orchestration
// ---------------------------------------------------------------------------
extern "C" void kernel_launch(void* const* d_in, const int* in_sizes, int n_in,
                              void* d_out, int out_size, void* d_ws, size_t ws_size,
                              hipStream_t stream)
{
    (void)in_sizes; (void)n_in; (void)out_size; (void)ws_size;

    const float* F1[3]  = { (const float*)d_in[4], (const float*)d_in[2], (const float*)d_in[0] };
    const float* F2[3]  = { (const float*)d_in[5], (const float*)d_in[3], (const float*)d_in[1] };
    const float* CTX[3] = { (const float*)d_in[8], (const float*)d_in[7], (const float*)d_in[6] };

    char* base = (char*)d_ws;
    size_t off = 0;
    auto alloc = [&](size_t bytes) -> void* {
        off = (off + 255) & ~(size_t)255;
        void* p = base + off;
        off += bytes;
        return p;
    };
    auto g1 = [](long n) { return dim3((unsigned)((n + 255) / 256)); };

    // ---- pack all weights to f16 (K padded to multiple of 64) ----
    auto packW = [&](int idx, int M, int K, int Kp) -> h16* {
        h16* dst = (h16*)alloc((size_t)M * Kp * sizeof(h16));
        long n = (long)M * Kp;
        pack_weight_f16<<<g1(n), 256, 0, stream>>>((const float*)d_in[idx], dst, M, K, Kp);
        return dst;
    };
    h16* w_mc1 = packW(9,  256,   81,  128);
    h16* w_mc2 = packW(11, 192, 2304, 2304);
    h16* w_mf1 = packW(13, 128,   98,  128);
    h16* w_mf2 = packW(15,  64, 1152, 1152);
    h16* w_mo  = packW(17, 126, 2304, 2304);
    h16* w_gz1 = packW(19, 128, 1920, 1920);
    h16* w_gr1 = packW(21, 128, 1920, 1920);
    h16* w_gq1 = packW(23, 128, 1920, 1920);
    h16* w_gz2 = packW(25, 128, 1920, 1920);
    h16* w_gr2 = packW(27, 128, 1920, 1920);
    h16* w_gq2 = packW(29, 128, 1920, 1920);
    h16* w_fh1 = packW(31, 256, 1152, 1152);
    h16* w_fh2 = packW(33,   2, 2304, 2304);
    h16* w_up1 = packW(35, 256, 1152, 1152);
    h16* w_up2 = packW(37, 576,  256,  256);
    const float* b_mc1 = (const float*)d_in[10];
    const float* b_mc2 = (const float*)d_in[12];
    const float* b_mf1 = (const float*)d_in[14];
    const float* b_mf2 = (const float*)d_in[16];
    const float* b_mo  = (const float*)d_in[18];
    const float* b_gz1 = (const float*)d_in[20];
    const float* b_gr1 = (const float*)d_in[22];
    const float* b_gq1 = (const float*)d_in[24];
    const float* b_gz2 = (const float*)d_in[26];
    const float* b_gr2 = (const float*)d_in[28];
    const float* b_gq2 = (const float*)d_in[30];
    const float* b_fh1 = (const float*)d_in[32];
    const float* b_fh2 = (const float*)d_in[34];
    const float* b_up1 = (const float*)d_in[36];
    const float* b_up2 = (const float*)d_in[38];

    // ---- workspace buffers (sized for finest level: HW=3072, N=6144) ----
    const int MAXHW = 48 * 64;
    const long MAXN = 2L * MAXHW;
    h16*   At      = (h16*)  alloc((size_t)MAXHW * 256 * sizeof(h16));
    h16*   Bt      = (h16*)  alloc((size_t)MAXHW * 256 * sizeof(h16));
    float* corrvol = (float*)alloc((size_t)2 * MAXHW * MAXHW * sizeof(float));
    h16*   col     = (h16*)  alloc((size_t)2304 * MAXN * sizeof(h16));
    float* h_buf   = (float*)alloc((size_t)128 * MAXN * sizeof(float));
    float* ctx_buf = (float*)alloc((size_t)128 * MAXN * sizeof(float));
    float* x_full  = (float*)alloc((size_t)256 * MAXN * sizeof(float));
    float* hx      = (float*)alloc((size_t)384 * MAXN * sizeof(float));
    float* zb      = (float*)alloc((size_t)128 * MAXN * sizeof(float));
    float* rb      = (float*)alloc((size_t)128 * MAXN * sizeof(float));
    float* qb      = (float*)alloc((size_t)128 * MAXN * sizeof(float));
    float* corrf   = (float*)alloc((size_t)81  * MAXN * sizeof(float));
    float* cor1    = (float*)alloc((size_t)256 * MAXN * sizeof(float)); // also fh1 out
    float* cor2    = (float*)alloc((size_t)192 * MAXN * sizeof(float));
    float* flo1    = (float*)alloc((size_t)128 * MAXN * sizeof(float));
    float* flo2    = (float*)alloc((size_t)64  * MAXN * sizeof(float));
    float* cf      = (float*)alloc((size_t)256 * MAXN * sizeof(float)); // also up1 out
    float* dbuf    = (float*)alloc((size_t)2   * MAXN * sizeof(float));
    float* up2o    = (float*)alloc((size_t)576 * MAXN * sizeof(float));
    float* flowA   = (float*)alloc((size_t)2   * MAXN * sizeof(float));
    float* flowB   = (float*)alloc((size_t)2   * MAXN * sizeof(float));

    auto gemm = [&](const h16* A, const h16* B, float* C, int M, int N, int K,
                    const float* bias, float alpha, int act) {
        dim3 grid((unsigned)((N + BLK_N - 1) / BLK_N), (unsigned)((M + BLK_M - 1) / BLK_M));
        gemm_wmma_f16<<<grid, 256, 0, stream>>>(A, B, C, M, N, K, bias, alpha, act);
    };

    const int dims[3][2] = { {12, 16}, {24, 32}, {48, 64} };
    const int iters[3]   = { 4, 3, 3 };

    float* flow = flowA;
    int prevH = 0, prevW = 0;
    long outoff = 0;
    float* out = (float*)d_out;

    for (int lvl = 0; lvl < 3; ++lvl) {
        const int H = dims[lvl][0], W = dims[lvl][1];
        const int HW = H * W;
        const long N = 2L * HW;          // B*H*W

        // implicit-GEMM conv on a [Cin, B, H, W] activation
        auto conv = [&](const float* src, int Cin, int kh, int kw, int Kp,
                        const h16* wgt, const float* bias, int Cout,
                        float* dst, int act) {
            long total = (long)Kp * N;
            im2col_f16<<<g1(total), 256, 0, stream>>>(src, col, Cin, 2, H, W, kh, kw, Kp);
            gemm(wgt, col, dst, Cout, (int)N, Kp, bias, 1.f, act);
        };

        // context -> h (tanh), ctx (relu); x rows 0..127 = ctx (constant per level)
        bchw_to_cbhw<<<g1(256 * N), 256, 0, stream>>>(CTX[lvl], x_full, 256, 2, HW);
        ctx_split<<<g1(256 * N), 256, 0, stream>>>(x_full, h_buf, ctx_buf, N);
        copy_f32k<<<g1(128 * N), 256, 0, stream>>>(x_full, ctx_buf, 128 * N);

        // flow init / upsample from previous level
        if (lvl == 0) {
            fill0_f32<<<g1(2 * N), 256, 0, stream>>>(flow, 2 * N);
        } else {
            float* nf = (flow == flowA) ? flowB : flowA;
            resize_flow<<<g1(2 * N), 256, 0, stream>>>(flow, nf, 2, prevH, prevW, H, W);
            flow = nf;
        }

        // all-pairs correlation volume per batch: [HW,256] x [256,HW]^T / 16
        for (int b = 0; b < 2; ++b) {
            pack_corr_f16<<<g1((long)HW * 256), 256, 0, stream>>>(F1[lvl] + (long)b * 256 * HW, At, HW);
            pack_corr_f16<<<g1((long)HW * 256), 256, 0, stream>>>(F2[lvl] + (long)b * 256 * HW, Bt, HW);
            gemm(At, Bt, corrvol + (long)b * HW * HW, HW, HW, 256, nullptr, 0.0625f, 0);
        }

        for (int it = 0; it < iters[lvl]; ++it) {
            // correlation lookup (81 taps)
            corr_lookup_kernel<<<g1(81 * N), 256, 0, stream>>>(corrvol, flow, corrf, 2, H, W);

            // motion encoder
            conv(corrf, 81, 1, 1,  128, w_mc1, b_mc1, 256, cor1, 1);
            conv(cor1, 256, 3, 3, 2304, w_mc2, b_mc2, 192, cor2, 1);
            conv(flow,   2, 7, 7,  128, w_mf1, b_mf1, 128, flo1, 1);
            conv(flo1, 128, 3, 3, 1152, w_mf2, b_mf2,  64, flo2, 1);
            copy_f32k<<<g1(192 * N), 256, 0, stream>>>(cf,            cor2, 192 * N);
            copy_f32k<<<g1(64  * N), 256, 0, stream>>>(cf + 192 * N,  flo2,  64 * N);
            conv(cf,   256, 3, 3, 2304, w_mo,  b_mo,  126, x_full + 128 * N, 1);
            copy_f32k<<<g1(2 * N), 256, 0, stream>>>(x_full + 254 * N, flow, 2 * N);

            // SepConvGRU pass 1 (1x5)
            copy_f32k<<<g1(128 * N), 256, 0, stream>>>(hx,            h_buf, 128 * N);
            copy_f32k<<<g1(256 * N), 256, 0, stream>>>(hx + 128 * N,  x_full, 256 * N);
            conv(hx, 384, 1, 5, 1920, w_gz1, b_gz1, 128, zb, 2);
            conv(hx, 384, 1, 5, 1920, w_gr1, b_gr1, 128, rb, 2);
            ewise_mul<<<g1(128 * N), 256, 0, stream>>>(hx, rb, h_buf, 128 * N);
            conv(hx, 384, 1, 5, 1920, w_gq1, b_gq1, 128, qb, 3);
            gru_combine<<<g1(128 * N), 256, 0, stream>>>(h_buf, zb, qb, 128 * N);

            // SepConvGRU pass 2 (5x1)
            copy_f32k<<<g1(128 * N), 256, 0, stream>>>(hx, h_buf, 128 * N);
            conv(hx, 384, 5, 1, 1920, w_gz2, b_gz2, 128, zb, 2);
            conv(hx, 384, 5, 1, 1920, w_gr2, b_gr2, 128, rb, 2);
            ewise_mul<<<g1(128 * N), 256, 0, stream>>>(hx, rb, h_buf, 128 * N);
            conv(hx, 384, 5, 1, 1920, w_gq2, b_gq2, 128, qb, 3);
            gru_combine<<<g1(128 * N), 256, 0, stream>>>(h_buf, zb, qb, 128 * N);

            // flow head
            conv(h_buf, 128, 3, 3, 1152, w_fh1, b_fh1, 256, cor1, 1);
            conv(cor1,  256, 3, 3, 2304, w_fh2, b_fh2,   2, dbuf, 0);
            add_inplace<<<g1(2 * N), 256, 0, stream>>>(flow, dbuf, 2 * N);

            // outputs
            if (lvl < 2) {
                flow_to_out<<<g1(2 * N), 256, 0, stream>>>(flow, out + outoff, 2, HW);
                outoff += 2 * N;
            } else {
                conv(h_buf, 128, 3, 3, 1152, w_up1, b_up1, 256, cf,   1);
                conv(cf,    256, 1, 1,  256, w_up2, b_up2, 576, up2o, 0);
                up8_kernel<<<g1(64 * 2 * N), 256, 0, stream>>>(up2o, flow, out + outoff, 2, H, W);
                outoff += 64 * 2 * N;   // B*2*(8H)*(8W)
            }
        }
        prevH = H; prevW = W;
    }
}
